// SelfAttention_31318901522697
// MI455X (gfx1250) — compile-verified, matrix operand order checked by reading, not run on hardware
//
#include <hip/hip_runtime.h>

// ---------------------------------------------------------------------------
// Self-attention (B=8, C=3840, N=1024) for gfx1250 via bf16 WMMA.
// cvt(fp32->bf16, pad) -> q/k/v GEMMs -> energy GEMM -> softmax -> out GEMM
// GEMM: 64x128 block tile, 8 waves (2x4), 32x32 per wave = 4 WMMA / K-step,
// LDS double-buffered, one barrier per K-step. Direct (row-major) tile
// staging uses GLOBAL_LOAD_ASYNC_TO_LDS_B128 (ASYNCcnt) to overlap with WMMA.
// ---------------------------------------------------------------------------

typedef unsigned short u16;
typedef __attribute__((ext_vector_type(16))) __bf16 v16bf;
typedef __attribute__((ext_vector_type(8)))  float  v8f;

union FragAB { v16bf v; unsigned u[8]; };
union FragC  { v8f  v; float f[8]; };

__device__ __forceinline__ u16 f2bf(float f) {
  unsigned u = __builtin_bit_cast(unsigned, f);
  u += 0x7FFFu + ((u >> 16) & 1u);          // round-to-nearest-even
  return (u16)(u >> 16);
}

// ---------------------------------------------------------------------------
// Stage an R x 32 bf16 tile into LDS laid out row-major [R][32] (row = M or N,
// col = K). KM=false: source row-major -> per-lane contiguous 16B copies,
// issued as async global->LDS (tracked by ASYNCcnt, waited before barrier).
// KM=true : source k-major, scatter via ds_store_b16. 256 threads cooperate.
// ---------------------------------------------------------------------------
template <int R, bool KM>
__device__ __forceinline__ void stage_tile(u16* lds, const u16* srcp, int ld, int t) {
  constexpr int E = (R * 32) / 256;               // bf16 elements per thread (8/16)
  constexpr int V = E / 8;                        // 16B chunks per thread
  if constexpr (!KM) {
    constexpr int TPR = 32 / E;                   // threads per row
    const int r = t / TPR;
    const int k = (t % TPR) * E;
#pragma unroll
    for (int c = 0; c < V; ++c) {
      const u16* s = srcp + (size_t)r * ld + k + c * 8;
      u16* d = lds + r * 32 + k + c * 8;
      const unsigned ldsoff = (unsigned)(size_t)d;  // low 32 bits = LDS address
      asm volatile("global_load_async_to_lds_b128 %0, %1, off"
                   :: "v"(ldsoff), "v"(s) : "memory");
    }
  } else {
    constexpr int TPK = R / E;                    // threads per k-line
    const int k = t / TPK;
    const int r = (t % TPK) * E;
#pragma unroll
    for (int c = 0; c < V; ++c) {
      u16 tmp[8];
      *(uint4*)tmp = *(const uint4*)(srcp + (size_t)k * ld + r + c * 8);
#pragma unroll
      for (int j = 0; j < 8; ++j) lds[(r + c * 8 + j) * 32 + k] = tmp[j];
    }
  }
}

// Load one 16x32 bf16 fragment from an LDS tile stored [row][k].
// CDNA5 ISA 7.12.2 16-bit A/B layout: lanes 0-15 = rows 0-15 with
// K={0..7,16..23}; lanes 16-31 = same rows with K={8..15,24..31};
// VGPR v packs K pair (base + 2*(v&3), +1)  -> two 16B LDS loads per frag.
__device__ __forceinline__ void load_frag(FragAB& f, const u16* lds, int rowbase, int lane) {
  const int hf  = lane >> 4;
  const int row = lane & 15;
  const u16* p = lds + (rowbase + row) * 32 + hf * 8;
#pragma unroll
  for (int v = 0; v < 8; ++v) {
    const int k0 = ((v < 4) ? 0 : 16) + (v & 3) * 2;
    f.u[v] = *(const unsigned*)(p + k0);
  }
}

// ---------------------------------------------------------------------------
// GEMM: Y[m,n] = sum_k A(m,k) * B(n,k) (B logical N x K). 64x128 block tile.
// EPI 0: y += bias[m] (bias pre-padded to M), store bf16 (row stride 1024)
// EPI 1: store fp32 (row stride 1024)
// EPI 2: y = y*gamma + x[idx], store fp32 (row stride 1024)
// ---------------------------------------------------------------------------
template <bool KMA, bool KMB, int EPI>
__global__ __launch_bounds__(256) void gemm64x128(
    const u16* __restrict__ Ag, long long strideA, int lda,
    const u16* __restrict__ Bg, long long strideB, int ldb,
    int K,
    const float* __restrict__ bias,
    u16*   __restrict__ Ybf, long long strideYb,
    float* __restrict__ Yf,  long long strideYf,
    const float* __restrict__ Xres, long long strideX,
    const float* __restrict__ gamma) {
  __shared__ __align__(16) u16 ldsA[2][64 * 32];
  __shared__ __align__(16) u16 ldsB[2][128 * 32];

  const int t    = threadIdx.x;
  const int lane = t & 31;
  const int wave = t >> 5;
  const int bm    = blockIdx.y * 64;
  const int bn    = blockIdx.x * 128;
  const int batch = blockIdx.z;

  const u16* Ab = Ag + (size_t)batch * strideA;
  const u16* Bb = Bg + (size_t)batch * strideB;

  const int wm = (wave >> 2) * 32;   // wave M offset inside tile (0/32)
  const int wn = (wave & 3) * 32;    // wave N offset inside tile (0..96)

  FragC acc[2][2];
#pragma unroll
  for (int i = 0; i < 2; ++i)
#pragma unroll
    for (int j = 0; j < 2; ++j)
#pragma unroll
      for (int g = 0; g < 8; ++g) acc[i][j].f[g] = 0.0f;

  auto stage = [&](int buf, int kb) {
    const u16* Ap = KMA ? (Ab + (size_t)kb * lda + bm) : (Ab + (size_t)bm * lda + kb);
    const u16* Bp = KMB ? (Bb + (size_t)kb * ldb + bn) : (Bb + (size_t)bn * ldb + kb);
    stage_tile<64,  KMA>(ldsA[buf], Ap, lda, t);
    stage_tile<128, KMB>(ldsB[buf], Bp, ldb, t);
  };
  // Direct staging issues async global->LDS; wait before publishing barrier.
  auto publish = [&]() {
    if constexpr (!KMA || !KMB)
      asm volatile("s_wait_asynccnt 0x0" ::: "memory");
    __syncthreads();
  };

  const int nk = K >> 5;
  stage(0, 0);
  publish();

  for (int kb = 0; kb < nk; ++kb) {
    const int cur = kb & 1;
    if (kb + 1 < nk) stage(cur ^ 1, (kb + 1) << 5);   // overlap with compute

    FragAB a0, a1, b0, b1;
    load_frag(a0, ldsA[cur], wm,      lane);
    load_frag(a1, ldsA[cur], wm + 16, lane);
    load_frag(b0, ldsB[cur], wn,      lane);
    load_frag(b1, ldsB[cur], wn + 16, lane);
    acc[0][0].v = __builtin_amdgcn_wmma_f32_16x16x32_bf16(
        false, a0.v, false, b0.v, (short)0, acc[0][0].v, false, false);
    acc[0][1].v = __builtin_amdgcn_wmma_f32_16x16x32_bf16(
        false, a0.v, false, b1.v, (short)0, acc[0][1].v, false, false);
    acc[1][0].v = __builtin_amdgcn_wmma_f32_16x16x32_bf16(
        false, a1.v, false, b0.v, (short)0, acc[1][0].v, false, false);
    acc[1][1].v = __builtin_amdgcn_wmma_f32_16x16x32_bf16(
        false, a1.v, false, b1.v, (short)0, acc[1][1].v, false, false);

    if (kb + 1 < nk) publish();
  }

  // Epilogue. C/D layout: VGPR g -> M = g + 8*(lane/16), N = lane%16.
  const int hf  = lane >> 4;
  const int col = lane & 15;
#pragma unroll
  for (int i = 0; i < 2; ++i)
#pragma unroll
    for (int j = 0; j < 2; ++j) {
      const int gm0 = bm + wm + i * 16 + hf * 8;
      const int gn  = bn + wn + j * 16 + col;
#pragma unroll
      for (int g = 0; g < 8; ++g) {
        const int gm = gm0 + g;
        float y = acc[i][j].f[g];
        if constexpr (EPI == 0) {
          y += bias[gm];
          Ybf[(size_t)batch * strideYb + (size_t)gm * 1024 + gn] = f2bf(y);
        } else if constexpr (EPI == 1) {
          Yf[(size_t)batch * strideYf + (size_t)gm * 1024 + gn] = y;
        } else {
          const size_t idx = (size_t)batch * strideX + (size_t)gm * 1024 + gn;
          Yf[idx] = y * gamma[0] + Xres[idx];
        }
      }
    }
}

// ---------------------------------------------------------------------------
// Row softmax: one block per row of 1024 fp32, emit bf16 attention weights.
// ---------------------------------------------------------------------------
__global__ __launch_bounds__(256) void softmax_row(const float* __restrict__ E,
                                                   u16* __restrict__ A) {
  const size_t row = blockIdx.x;
  const float* e = E + row * 1024;
  u16* a = A + row * 1024;
  const int t = threadIdx.x;
  const int lane = t & 31;
  const int wave = t >> 5;
  __shared__ float red[8];

  float v[4];
  float m = -3.4e38f;
#pragma unroll
  for (int i = 0; i < 4; ++i) { v[i] = e[t + i * 256]; m = fmaxf(m, v[i]); }
#pragma unroll
  for (int o = 16; o > 0; o >>= 1) m = fmaxf(m, __shfl_xor(m, o, 32));
  if (lane == 0) red[wave] = m;
  __syncthreads();
  if (wave == 0) {
    float mm = red[lane & 7];
#pragma unroll
    for (int o = 4; o > 0; o >>= 1) mm = fmaxf(mm, __shfl_xor(mm, o, 32));
    if (lane == 0) red[0] = mm;
  }
  __syncthreads();
  m = red[0];
  __syncthreads();

  float s = 0.0f;
#pragma unroll
  for (int i = 0; i < 4; ++i) { v[i] = __expf(v[i] - m); s += v[i]; }
#pragma unroll
  for (int o = 16; o > 0; o >>= 1) s += __shfl_xor(s, o, 32);
  if (lane == 0) red[wave] = s;
  __syncthreads();
  if (wave == 0) {
    float ss = red[lane & 7];
#pragma unroll
    for (int o = 4; o > 0; o >>= 1) ss += __shfl_xor(ss, o, 32);
    if (lane == 0) red[0] = ss;
  }
  __syncthreads();
  const float inv = 1.0f / red[0];
#pragma unroll
  for (int i = 0; i < 4; ++i) a[t + i * 256] = f2bf(v[i] * inv);
}

// fp32 -> bf16 convert with zero tail padding: pairs [0,nsrc) converted,
// pairs [nsrc,ndst) written as zero.
__global__ __launch_bounds__(256) void cvt_bf16(const float* __restrict__ in,
                                                unsigned* __restrict__ out,
                                                int nsrc, int ndst) {
  const int i = blockIdx.x * 256 + threadIdx.x;
  if (i < nsrc) {
    const float2 f = ((const float2*)in)[i];
    out[i] = (unsigned)f2bf(f.x) | ((unsigned)f2bf(f.y) << 16);
  } else if (i < ndst) {
    out[i] = 0u;
  }
}

// Pad a bias vector with zeros up to np elements (straight-line epilogues).
__global__ __launch_bounds__(256) void pad_bias(const float* __restrict__ b,
                                                float* __restrict__ o,
                                                int n, int np) {
  const int i = blockIdx.x * 256 + threadIdx.x;
  if (i < np) o[i] = (i < n) ? b[i] : 0.0f;
}

// ---------------------------------------------------------------------------
extern "C" void kernel_launch(void* const* d_in, const int* in_sizes, int n_in,
                              void* d_out, int out_size, void* d_ws, size_t ws_size,
                              hipStream_t stream) {
  (void)in_sizes; (void)n_in; (void)out_size; (void)ws_size;
  constexpr int Bn = 8, C = 3840, C8 = 480, C8P = 512, Nn = 1024;

  const float* x     = (const float*)d_in[0];
  const float* Wq    = (const float*)d_in[1];
  const float* bq    = (const float*)d_in[2];
  const float* Wk    = (const float*)d_in[3];
  const float* bk    = (const float*)d_in[4];
  const float* Wv    = (const float*)d_in[5];
  const float* bv    = (const float*)d_in[6];
  const float* gamma = (const float*)d_in[7];
  float* out = (float*)d_out;

  char* ws = (char*)d_ws;
  size_t off = 0;
  auto alloc_bf = [&](size_t elems) { u16* p = (u16*)(ws + off); off += elems * 2; return p; };
  u16* xb    = alloc_bf((size_t)Bn * C * Nn);     // x bf16              (B,C,N)
  u16* wqb   = alloc_bf((size_t)C8P * C);         // Wq bf16, zero-padded to 512 rows
  u16* wkb   = alloc_bf((size_t)C8P * C);
  u16* wvb   = alloc_bf((size_t)C * C);
  u16* qb    = alloc_bf((size_t)Bn * C8P * Nn);   // q bf16              (B,512,N)
  u16* kbuf  = alloc_bf((size_t)Bn * C8P * Nn);   // k bf16
  u16* vb    = alloc_bf((size_t)Bn * C * Nn);     // v bf16
  u16* attnb = alloc_bf((size_t)Bn * Nn * Nn);    // softmax(E) bf16     (B,N,N)
  float* energy = (float*)(ws + off); off += (size_t)Bn * Nn * Nn * 4;
  float* bqp = (float*)(ws + off); off += C8P * 4; // padded biases (512 f32)
  float* bkp = (float*)(ws + off); off += C8P * 4;

  auto cvt = [&](const float* src, u16* dst, size_t nsrc, size_t ndst) {
    const int ps = (int)(nsrc / 2), pd = (int)(ndst / 2);
    cvt_bf16<<<(pd + 255) / 256, 256, 0, stream>>>(src, (unsigned*)dst, ps, pd);
  };
  cvt(x,  xb,  (size_t)Bn * C * Nn, (size_t)Bn * C * Nn);
  cvt(Wq, wqb, (size_t)C8 * C, (size_t)C8P * C);
  cvt(Wk, wkb, (size_t)C8 * C, (size_t)C8P * C);
  cvt(Wv, wvb, (size_t)C * C, (size_t)C * C);
  pad_bias<<<2, 256, 0, stream>>>(bq, bqp, C8, C8P);
  pad_bias<<<2, 256, 0, stream>>>(bk, bkp, C8, C8P);

  // q/k/v projections: A = W (Opad x C row-major), B = x (C x N, k-major).
  gemm64x128<false, true, 0><<<dim3(Nn / 128, C8P / 64, Bn), 256, 0, stream>>>(
      wqb, 0LL, C, xb, (long long)C * Nn, Nn, C, bqp,
      qb, (long long)C8P * Nn, nullptr, 0LL, nullptr, 0LL, nullptr);
  gemm64x128<false, true, 0><<<dim3(Nn / 128, C8P / 64, Bn), 256, 0, stream>>>(
      wkb, 0LL, C, xb, (long long)C * Nn, Nn, C, bkp,
      kbuf, (long long)C8P * Nn, nullptr, 0LL, nullptr, 0LL, nullptr);
  gemm64x128<false, true, 0><<<dim3(Nn / 128, C / 64, Bn), 256, 0, stream>>>(
      wvb, 0LL, C, xb, (long long)C * Nn, Nn, C, bv,
      vb, (long long)C * Nn, nullptr, 0LL, nullptr, 0LL, nullptr);

  // energy[i,j] = sum_c q[c,i]*k[c,j]; K=512 (padded rows contribute zero).
  gemm64x128<true, true, 1><<<dim3(Nn / 128, Nn / 64, Bn), 256, 0, stream>>>(
      qb, (long long)C8P * Nn, Nn, kbuf, (long long)C8P * Nn, Nn, C8P, nullptr,
      nullptr, 0LL, energy, (long long)Nn * Nn, nullptr, 0LL, nullptr);

  // softmax over j for each (b,i) row
  softmax_row<<<Bn * Nn, 256, 0, stream>>>(energy, attnb);

  // out[c,i] = sum_j v[c,j]*attn[i,j]; fused *gamma + x epilogue into d_out.
  gemm64x128<false, false, 2><<<dim3(Nn / 128, C / 64, Bn), 256, 0, stream>>>(
      vb, (long long)C * Nn, Nn, attnb, (long long)Nn * Nn, Nn, Nn, nullptr,
      nullptr, 0LL, out, 0LL, x, (long long)C * Nn, gamma);
}
